// PlgaLayer_63196148793962
// MI455X (gfx1250) — matrix operand order repaired and checked by reading, not compiled, not run on hardware
//
#include <hip/hip_runtime.h>

#define Bdim 2
#define Hdim 16
#define Sdim 2048
#define Ddim 64
#define NEG_BIG (-1.0e9f)

typedef __attribute__((ext_vector_type(16))) __bf16 v16bf;
typedef __attribute__((ext_vector_type(8)))  float  v8f;

union AFrag { v16bf v; unsigned int d[8]; uint4 q[2]; };

#if defined(__gfx1250__) && __has_builtin(__builtin_amdgcn_tensor_load_to_lds)
#define USE_TDM 1
#else
#define USE_TDM 0
#endif

__device__ __forceinline__ unsigned short f2bf(float f) {
    union { float f; unsigned int u; } c; c.f = f;
    unsigned int u = c.u + 0x7FFFu + ((c.u >> 16) & 1u);
    return (unsigned short)(u >> 16);
}

// pack two floats to packed bf16 dword via v_perm_b32
__device__ __forceinline__ unsigned int pack_bf16(float lo, float hi) {
    union { float f; unsigned int u; } a, b; a.f = lo; b.f = hi;
    unsigned int ra = a.u + 0x7FFFu + ((a.u >> 16) & 1u);
    unsigned int rb = b.u + 0x7FFFu + ((b.u >> 16) & 1u);
    return __builtin_amdgcn_perm(rb, ra, 0x07060302u);   // {rb[31:16], ra[31:16]}
}

#if USE_TDM
typedef __attribute__((ext_vector_type(4))) unsigned int tg0_t;
typedef __attribute__((ext_vector_type(8))) int          tg1_t;
typedef __attribute__((ext_vector_type(4))) int          tg2_t;

// 2D bf16 tile load: 64 rows x 64 elems, global row stride = stride_elems,
// LDS rows padded to 72 elems (pad_interval=32 DWORDs, pad_amount=4 DWORDs)
__device__ __forceinline__ void tdm_load_tile(unsigned int lds_off, const void* gptr,
                                              unsigned int stride_elems) {
    unsigned long long ga = (unsigned long long)gptr;
    tg0_t g0;
    g0[0] = 1u;                                   // count=1, user descriptor
    g0[1] = lds_off;                              // lds_addr
    g0[2] = (unsigned int)ga;                     // global_addr[31:0]
    g0[3] = (unsigned int)(ga >> 32) | 0x80000000u; // global_addr[56:32] | type=2
    tg1_t g1;
    g1[0] = (int)((1u << 16) | (1u << 20) | (4u << 22) | (3u << 25)); // 2B, pad on, 32dw interval, 4dw pad
    g1[1] = (int)(64u << 16);                     // tensor_dim0 = 64
    g1[2] = (int)(64u << 16);                     // tensor_dim1 = 64
    g1[3] = (int)(64u << 16);                     // tile_dim0 = 64
    g1[4] = 64;                                   // tile_dim1 = 64
    g1[5] = (int)stride_elems;                    // tensor_dim0_stride
    g1[6] = 0; g1[7] = 0;
    tg2_t z4 = {0, 0, 0, 0};
    tg1_t z8 = {0, 0, 0, 0, 0, 0, 0, 0};
    __builtin_amdgcn_tensor_load_to_lds(g0, g1, z4, z4, z8, 0);  // 6-arg form (clang-23)
}
#endif

// ---------------------------------------------------------------- K->bf16, V->bf16 transposed [b,h,d,s]
__global__ __launch_bounds__(256) void cvt_kernel(const float* __restrict__ K,
                                                  const float* __restrict__ V,
                                                  unsigned short* __restrict__ Kb,
                                                  unsigned short* __restrict__ Vb) {
    const int bh = blockIdx.x, s0 = blockIdx.y * 64;
    __shared__ unsigned short T[64 * 66];
    const float* Ks = K + ((size_t)bh * Sdim + s0) * 64;
    const float* Vs = V + ((size_t)bh * Sdim + s0) * 64;
    for (int e = threadIdx.x; e < 4096; e += 256) {
        int r = e >> 6, d = e & 63;
        Kb[((size_t)bh * Sdim + s0 + r) * 64 + d] = f2bf(Ks[e]);
        T[d * 66 + r] = f2bf(Vs[e]);
    }
    __syncthreads();
    for (int e = threadIdx.x; e < 4096; e += 256) {
        int d = e >> 6, r = e & 63;
        Vb[((size_t)bh * 64 + d) * Sdim + s0 + r] = T[d * 66 + r];
    }
}

// ---------------------------------------------------------------- G tensor
__global__ __launch_bounds__(256) void g_kernel(const float* __restrict__ A,
                                                const float* __restrict__ W,
                                                const float* __restrict__ bb,
                                                const float* __restrict__ pw,
                                                const float* __restrict__ av,
                                                const float* __restrict__ ba,
                                                float* __restrict__ G) {
    const int bh = blockIdx.x;
    const int h  = bh % Hdim;
    __shared__ float As[64 * 65];
    __shared__ float Ap[64 * 65];
    const float* Abh = A + (size_t)bh * 4096;
    const float* Wh  = W + (size_t)h * 4096;
    for (int e = threadIdx.x; e < 4096; e += 256)
        As[(e >> 6) * 65 + (e & 63)] = Abh[e];
    __syncthreads();
    for (int e = threadIdx.x; e < 4096; e += 256) {
        int i = e >> 6, k = e & 63;
        float acc = bb[(size_t)h * 4096 + e];
        #pragma unroll 8
        for (int j = 0; j < 64; ++j) acc += Wh[i * 64 + j] * As[j * 65 + k];
        float sig = 1.0f / (1.0f + __expf(-acc));
        float aw  = acc * acc * sig + 1e-9f;
        Ap[i * 65 + k] = __powf(aw, pw[(size_t)h * 4096 + e]);
    }
    __syncthreads();
    for (int e = threadIdx.x; e < 4096; e += 256) {
        int i = e >> 6, k = e & 63;
        float acc = ba[(size_t)h * 4096 + e];
        const float* avr = av + (size_t)h * 4096 + i * 64;
        #pragma unroll 8
        for (int j = 0; j < 64; ++j) acc += avr[j] * Ap[j * 65 + k];
        G[(size_t)bh * 4096 + e] = acc;
    }
}

// ---------------------------------------------------------------- Q = (Hin @ G) * log2(e)/sqrt(D), bf16
__global__ __launch_bounds__(256) void q_kernel(const float* __restrict__ Hin,
                                                const float* __restrict__ G,
                                                unsigned short* __restrict__ Qb) {
    const int bh = blockIdx.x, chunk = blockIdx.y;
    __shared__ float Gl[64 * 65];
    __shared__ float Hl[64 * 65];
    const float* Gs = G + (size_t)bh * 4096;
    const float* Hs = Hin + ((size_t)bh * Sdim + chunk * 64) * 64;
    for (int e = threadIdx.x; e < 4096; e += 256) {
        Gl[(e >> 6) * 65 + (e & 63)] = Gs[e];
        Hl[(e >> 6) * 65 + (e & 63)] = Hs[e];
    }
    __syncthreads();
    for (int e = threadIdx.x; e < 4096; e += 256) {
        int r = e >> 6, c = e & 63;
        float acc = 0.0f;
        #pragma unroll 8
        for (int j = 0; j < 64; ++j) acc += Hl[r * 65 + j] * Gl[j * 65 + c];
        Qb[((size_t)bh * Sdim + chunk * 64 + r) * 64 + c] =
            f2bf(acc * (0.125f * 1.44269504088896f));   // exp2 domain
    }
}

// ---------------------------------------------------------------- flash attention (S^T = K Q^T, O^T = V^T P^T)
__global__ __launch_bounds__(256) void fa_kernel(const unsigned short* __restrict__ Qb,
                                                 const unsigned short* __restrict__ Kb,
                                                 const unsigned short* __restrict__ Vb,
                                                 float* __restrict__ out) {
    const int bh   = blockIdx.x;
    const int qblk = blockIdx.y;                  // 128 query rows per block
    const int b = bh / Hdim, h = bh % Hdim;
    const int tid  = threadIdx.x;
    const int wave = tid >> 5, lane = tid & 31;
    const int half = lane >> 4, ln = lane & 15;
    const int rowbase = qblk * 128 + wave * 16;
    const int myq = rowbase + ln;

    __shared__ __align__(16) union ShMem {
        struct {
            unsigned short K[2][64 * 72];         // [key][d], pitch 72
            unsigned short V[2][64 * 72];         // [d][key], pitch 72
            unsigned short P[8][16 * 72];         // per wave [query][key]
        } t;
        float O[8][16 * 68];
    } sh;

    // ---- Q B-fragments (d x queries) in registers
    AFrag qf[2];
    {
        const unsigned short* qrow = Qb + ((size_t)bh * Sdim + myq) * 64;
        #pragma unroll
        for (int c = 0; c < 2; ++c) {
            qf[c].q[0] = *(const uint4*)(qrow + c * 32 + 8 * half);
            qf[c].q[1] = *(const uint4*)(qrow + c * 32 + 8 * half + 16);
        }
    }

    v8f o0 = {}, o1 = {}, o2 = {}, o3 = {};
    float m = -1e30f, l = 0.0f;

    const int ntiles = 2 * qblk + 2;
    const unsigned short* Kbase = Kb + (size_t)bh * Sdim * 64;   // [s][d]
    const unsigned short* Vbase = Vb + (size_t)bh * 64 * Sdim;   // [d][s]

#if USE_TDM
    if (wave == 0) {
        tdm_load_tile((unsigned int)(size_t)&sh.t.K[0][0], Kbase, 64u);
        tdm_load_tile((unsigned int)(size_t)&sh.t.V[0][0], Vbase, (unsigned int)Sdim);
    }
#endif

    for (int t = 0; t < ntiles; ++t) {
        const int kb = t * 64;
#if USE_TDM
        if (wave == 0) __builtin_amdgcn_s_wait_tensorcnt(0);
        __syncthreads();                          // tile t visible; tile t-1 readers done
        if (wave == 0 && t + 1 < ntiles) {
            int nb = (t + 1) & 1;
            tdm_load_tile((unsigned int)(size_t)&sh.t.K[nb][0],
                          Kbase + (size_t)(kb + 64) * 64, 64u);
            tdm_load_tile((unsigned int)(size_t)&sh.t.V[nb][0],
                          Vbase + (size_t)(kb + 64), (unsigned int)Sdim);
        }
        const unsigned short* Kc = sh.t.K[t & 1];
        const unsigned short* Vc = sh.t.V[t & 1];
#else
        __syncthreads();
        {
            const unsigned short* Ksrc = Kbase + (size_t)kb * 64;
            const unsigned short* Vsrc = Vbase + kb;
            #pragma unroll
            for (int i0 = 0; i0 < 2; ++i0) {
                int i = tid + i0 * 256;           // 512 uint4 per tile
                int rr = i >> 3, e0 = (i & 7) * 8;
                uint4 kq = *(const uint4*)(Ksrc + rr * 64 + e0);
                *(uint4*)&sh.t.K[0][rr * 72 + e0] = kq;
                uint4 vq = *(const uint4*)(Vsrc + (size_t)rr * Sdim + e0);
                *(uint4*)&sh.t.V[0][rr * 72 + e0] = vq;
            }
            if (t + 1 < ntiles) __builtin_prefetch(Ksrc + 4096 + tid * 16, 0, 1);
        }
        __syncthreads();
        const unsigned short* Kc = sh.t.K[0];
        const unsigned short* Vc = sh.t.V[0];
#endif
        if (kb > rowbase + 15) continue;          // uniform per wave

        // ---- S^T = K Q^T : 8 WMMA
        v8f s0 = {}, s1 = {}, s2 = {}, s3 = {};
        #pragma unroll
        for (int c = 0; c < 2; ++c) {
            #pragma unroll
            for (int g = 0; g < 4; ++g) {
                AFrag kf;
                const unsigned short* kro = &Kc[(g * 16 + ln) * 72 + c * 32 + 8 * half];
                kf.q[0] = *(const uint4*)(kro);
                kf.q[1] = *(const uint4*)(kro + 16);
                if (g == 0) s0 = __builtin_amdgcn_wmma_f32_16x16x32_bf16(false, kf.v, false, qf[c].v, (short)0, s0, false, false);
                if (g == 1) s1 = __builtin_amdgcn_wmma_f32_16x16x32_bf16(false, kf.v, false, qf[c].v, (short)0, s1, false, false);
                if (g == 2) s2 = __builtin_amdgcn_wmma_f32_16x16x32_bf16(false, kf.v, false, qf[c].v, (short)0, s2, false, false);
                if (g == 3) s3 = __builtin_amdgcn_wmma_f32_16x16x32_bf16(false, kf.v, false, qf[c].v, (short)0, s3, false, false);
            }
        }

        // ---- additive causal mask
        if (kb + 63 > rowbase) {
            #pragma unroll
            for (int r = 0; r < 8; ++r) {
                int kr = kb + r + 8 * half;
                s0[r] += (kr      > myq) ? NEG_BIG : 0.0f;
                s1[r] += (kr + 16 > myq) ? NEG_BIG : 0.0f;
                s2[r] += (kr + 32 > myq) ? NEG_BIG : 0.0f;
                s3[r] += (kr + 48 > myq) ? NEG_BIG : 0.0f;
            }
        }

        // ---- online softmax in exp2 domain, one cross-half shuffle per stat
        float vmax = -1e30f;
        #pragma unroll
        for (int r = 0; r < 8; ++r)
            vmax = fmaxf(vmax, fmaxf(fmaxf(s0[r], s1[r]), fmaxf(s2[r], s3[r])));
        vmax = fmaxf(vmax, __shfl_xor(vmax, 16, 32));
        float mnew  = fmaxf(m, vmax);
        float scale = __builtin_amdgcn_exp2f(m - mnew);
        unsigned short* Pw = sh.t.P[wave];
        float ps = 0.0f;
        #pragma unroll
        for (int r = 0; r < 8; r += 2) {          // packed bf16 P stores
            float pa0 = __builtin_amdgcn_exp2f(s0[r] - mnew), pb0 = __builtin_amdgcn_exp2f(s0[r + 1] - mnew);
            float pa1 = __builtin_amdgcn_exp2f(s1[r] - mnew), pb1 = __builtin_amdgcn_exp2f(s1[r + 1] - mnew);
            float pa2 = __builtin_amdgcn_exp2f(s2[r] - mnew), pb2 = __builtin_amdgcn_exp2f(s2[r + 1] - mnew);
            float pa3 = __builtin_amdgcn_exp2f(s3[r] - mnew), pb3 = __builtin_amdgcn_exp2f(s3[r + 1] - mnew);
            ps += ((pa0 + pb0) + (pa1 + pb1)) + ((pa2 + pb2) + (pa3 + pb3));
            int kr = 8 * half + r;
            *(unsigned int*)&Pw[ln * 72 + kr]      = pack_bf16(pa0, pb0);
            *(unsigned int*)&Pw[ln * 72 + kr + 16] = pack_bf16(pa1, pb1);
            *(unsigned int*)&Pw[ln * 72 + kr + 32] = pack_bf16(pa2, pb2);
            *(unsigned int*)&Pw[ln * 72 + kr + 48] = pack_bf16(pa3, pb3);
        }
        ps += __shfl_xor(ps, 16, 32);
        l = l * scale + ps;
        m = mnew;
        #pragma unroll
        for (int r = 0; r < 8; ++r) { o0[r] *= scale; o1[r] *= scale; o2[r] *= scale; o3[r] *= scale; }

        // ---- O^T += V^T P^T : 8 WMMA
        AFrag pf[2];
        #pragma unroll
        for (int c = 0; c < 2; ++c) {
            const unsigned short* pro = &Pw[ln * 72 + c * 32 + 8 * half];
            pf[c].q[0] = *(const uint4*)(pro);
            pf[c].q[1] = *(const uint4*)(pro + 16);
        }
        #pragma unroll
        for (int c = 0; c < 2; ++c) {
            #pragma unroll
            for (int g = 0; g < 4; ++g) {
                AFrag vf;
                const unsigned short* vro = &Vc[(g * 16 + ln) * 72 + c * 32 + 8 * half];
                vf.q[0] = *(const uint4*)(vro);
                vf.q[1] = *(const uint4*)(vro + 16);
                if (g == 0) o0 = __builtin_amdgcn_wmma_f32_16x16x32_bf16(false, vf.v, false, pf[c].v, (short)0, o0, false, false);
                if (g == 1) o1 = __builtin_amdgcn_wmma_f32_16x16x32_bf16(false, vf.v, false, pf[c].v, (short)0, o1, false, false);
                if (g == 2) o2 = __builtin_amdgcn_wmma_f32_16x16x32_bf16(false, vf.v, false, pf[c].v, (short)0, o2, false, false);
                if (g == 3) o3 = __builtin_amdgcn_wmma_f32_16x16x32_bf16(false, vf.v, false, pf[c].v, (short)0, o3, false, false);
            }
        }
    }

    // ---- epilogue: O^T -> LDS -> coalesced float4 stores of Hout [B,S,H,D]
    __syncthreads();
    {
        float inv = 1.0f / l;
        float* Ow = sh.O[wave];
        #pragma unroll
        for (int r = 0; r < 8; ++r) {
            int dd = r + 8 * half;
            Ow[ln * 68 + dd]      = o0[r] * inv;
            Ow[ln * 68 + dd + 16] = o1[r] * inv;
            Ow[ln * 68 + dd + 32] = o2[r] * inv;
            Ow[ln * 68 + dd + 48] = o3[r] * inv;
        }
    }
    __syncthreads();
    {
        int qr = lane >> 1, ch = lane & 1;
        int row = rowbase + qr;
        float* orow = out + (((size_t)b * Sdim + row) * Hdim + h) * 64 + ch * 32;
        const float* src = &sh.O[wave][qr * 68 + ch * 32];
        #pragma unroll
        for (int j = 0; j < 8; ++j)
            *(float4*)(orow + j * 4) = *(const float4*)(src + j * 4);
    }
}

// ---------------------------------------------------------------- launcher
extern "C" void kernel_launch(void* const* d_in, const int* in_sizes, int n_in,
                              void* d_out, int out_size, void* d_ws, size_t ws_size,
                              hipStream_t stream) {
    const float* Hin  = (const float*)d_in[0];
    const float* Hk   = (const float*)d_in[1];
    const float* Hv   = (const float*)d_in[2];
    const float* A    = (const float*)d_in[3];
    // d_in[4] = mask (causal, analytic)
    const float* W    = (const float*)d_in[5];
    const float* bb   = (const float*)d_in[6];
    const float* pw   = (const float*)d_in[7];
    const float* av   = (const float*)d_in[8];
    const float* ba   = (const float*)d_in[9];
    float* out = (float*)d_out;

    const size_t N = (size_t)Bdim * Hdim * Sdim * Ddim;
    unsigned short* Qb = (unsigned short*)d_ws;
    unsigned short* Kb = Qb + N;
    unsigned short* Vb = Kb + N;
    float*          Gw = (float*)(Vb + N);

    cvt_kernel<<<dim3(Bdim * Hdim, Sdim / 64), 256, 0, stream>>>(Hk, Hv, Kb, Vb);
    g_kernel<<<Bdim * Hdim, 256, 0, stream>>>(A, W, bb, pw, av, ba, Gw);
    q_kernel<<<dim3(Bdim * Hdim, Sdim / 64), 256, 0, stream>>>(Hin, Gw, Qb);
    fa_kernel<<<dim3(Bdim * Hdim, Sdim / 128), 256, 0, stream>>>(Qb, Kb, Vb, out);
}
